// FrameTransformerEncoder_18013092839848
// MI455X (gfx1250) — matrix operand order repaired and checked by reading, not compiled, not run on hardware
//
#include <hip/hip_runtime.h>
#include <hip/hip_bf16.h>

typedef __bf16 bf16;
typedef __attribute__((ext_vector_type(16))) __bf16 v16bf;
typedef __attribute__((ext_vector_type(8)))  float   v8f;

union Frag { v16bf v; uint4 q[2]; };
union Acc8 { v8f   v; float f[8]; };

constexpr int Bn = 2, Cc = 8, FEAT = 1024, Wn = 512, HEADS = 8, Dh = 128, EXPF = 4096;

// ---------------------------------------------------------------------------
// async global->LDS copy of 32 bytes (2 x B128), tracked by ASYNCcnt.
// the instruction offset is added to both the LDS and the global address
// ---------------------------------------------------------------------------
__device__ __forceinline__ void async_copy32(const bf16* g, bf16* l) {
  unsigned lds = (unsigned)(uintptr_t)l;  // low 32 bits of generic LDS ptr = LDS byte addr
  asm volatile("global_load_async_to_lds_b128 %0, %1, off\n\t"
               "global_load_async_to_lds_b128 %0, %1, off offset:16"
               :: "v"(lds), "v"(g) : "memory");
}
__device__ __forceinline__ void async_wait0() {
  asm volatile("s_wait_asynccnt 0x0" ::: "memory");
}

// ---------------------------------------------------------------------------
// 1) embed: y[b,d,f,w] = sum_c x[b,c,f,w]*ew[d,c] + eb[d]
// ---------------------------------------------------------------------------
__global__ void embed_kernel(const float* __restrict__ x, const float* __restrict__ ew,
                             const float* __restrict__ eb, float* __restrict__ y) {
  const int FW = FEAT * Wn;
  int idx = blockIdx.x * blockDim.x + threadIdx.x;
  if (idx >= Bn * FW) return;
  int b = idx / FW, fw = idx % FW;
  float x0 = x[(size_t)(b * 2 + 0) * FW + fw];
  float x1 = x[(size_t)(b * 2 + 1) * FW + fw];
#pragma unroll
  for (int d = 0; d < Cc; ++d)
    y[(size_t)(b * Cc + d) * FW + fw] = x0 * ew[d * 2] + x1 * ew[d * 2 + 1] + eb[d];
}

// ---------------------------------------------------------------------------
// 2) frame_norm over FEAT axis; writes transposed bf16 [b,c,w,f] (K-contiguous)
// ---------------------------------------------------------------------------
__global__ void framenorm_t_kernel(const float* __restrict__ x,
                                   const float* __restrict__ gamma,
                                   const float* __restrict__ beta,
                                   bf16* __restrict__ yt) {
  int col = blockIdx.x * blockDim.x + threadIdx.x;  // over B*C*W
  if (col >= Bn * Cc * Wn) return;
  int bc = col / Wn, w = col % Wn;
  int c = bc % Cc;
  const float* xp = x + (size_t)bc * FEAT * Wn + w;
  float s = 0.f, s2 = 0.f;
  for (int f = 0; f < FEAT; ++f) {
    float v = xp[(size_t)f * Wn];
    s += v; s2 += v * v;
  }
  float mean = s * (1.f / FEAT);
  float var  = s2 * (1.f / FEAT) - mean * mean;
  float rstd = rsqrtf(var + 1e-5f);
  bf16* yp = yt + (size_t)bc * Wn * FEAT + (size_t)w * FEAT;
  const float* gp = gamma + (size_t)c * FEAT;
  const float* bp = beta  + (size_t)c * FEAT;
  for (int f = 0; f < FEAT; ++f) {
    float v = (xp[(size_t)f * Wn] - mean) * rstd * gp[f] + bp[f];
    yp[f] = (bf16)v;
  }
}

// ---------------------------------------------------------------------------
// 3) Generic batched NT GEMM: D[m,n] = scale * sum_k A[m,k]*B[n,k]  (bf16 WMMA)
//    A: f32 (converted on the fly, VALU stage) or bf16 (async-to-LDS stage);
//    B: bf16 (async-to-LDS stage), both K-contiguous.
//    Epilogue: bf16 store OR f32 store with optional residual add.
//    Block: 128x128 tile, 256 thr (8 waves, 4Mx2N), per-wave 2x4 WMMA tiles.
// ---------------------------------------------------------------------------
__launch_bounds__(256)
__global__ void gemm_nt_bf16_kernel(
    const void* __restrict__ Abase, int a_f32,
    const bf16* __restrict__ Bbase,
    float* __restrict__ dstF, const float* __restrict__ resF, bf16* __restrict__ dstB,
    int M, int N, int K,
    long long aStride, int aMod, long long bStride,
    long long dOuter, int dDiv, long long dInner, int ldd, float scale) {
  __shared__ __align__(16) bf16 As[2][128][40];
  __shared__ __align__(16) bf16 Bs[2][128][40];

  const int z  = blockIdx.z;
  const int za = aMod ? (z % aMod) : z;
  const float* Af = (const float*)Abase + (size_t)za * (size_t)aStride;
  const bf16*  Ah = (const bf16*) Abase + (size_t)za * (size_t)aStride;
  const bf16*  Bp = Bbase + (size_t)z * (size_t)bStride;
  const size_t doff = (size_t)(z / dDiv) * (size_t)dOuter + (size_t)(z % dDiv) * (size_t)dInner;

  const int tileM = blockIdx.y * 128;
  const int tileN = blockIdx.x * 128;

  const int tid  = threadIdx.x;
  const int srow = tid >> 1;          // 0..127 staging row
  const int scol = (tid & 1) * 16;    // 0 or 16 (bf16 elems within 32-wide k-slab)

  const int lane = tid & 31;
  const int wave = tid >> 5;
  const int wm = (wave & 3) * 32;     // wave M offset (4 M-waves)
  const int wn = (wave >> 2) * 64;    // wave N offset (2 N-waves)
  const int lr = lane & 15;
  const int hi = lane >> 4;

  Acc8 acc[2][4];
#pragma unroll
  for (int i = 0; i < 2; ++i)
#pragma unroll
    for (int j = 0; j < 4; ++j)
#pragma unroll
      for (int r = 0; r < 8; ++r) acc[i][j].f[r] = 0.f;

  auto stage = [&](int buf, int kk) {
    if (a_f32) {
      // f32 weights: load + convert to bf16 in VGPRs, ds-store to LDS
      const float* g = Af + (size_t)(tileM + srow) * K + kk + scol;
      __builtin_prefetch(g + 32, 0, 1);
      union { bf16 h[16]; uint4 u[2]; } t;
      const float4* g4 = (const float4*)g;
#pragma unroll
      for (int q = 0; q < 4; ++q) {
        float4 f4 = g4[q];
        t.h[4 * q + 0] = (bf16)f4.x; t.h[4 * q + 1] = (bf16)f4.y;
        t.h[4 * q + 2] = (bf16)f4.z; t.h[4 * q + 3] = (bf16)f4.w;
      }
      *(uint4*)(&As[buf][srow][scol])     = t.u[0];
      *(uint4*)(&As[buf][srow][scol] + 8) = t.u[1];
    } else {
      // bf16 A: async DMA straight into LDS (no VGPR round trip)
      async_copy32(Ah + (size_t)(tileM + srow) * K + kk + scol,
                   &As[buf][srow][scol]);
    }
    // B is always bf16: async DMA into LDS
    async_copy32(Bp + (size_t)(tileN + srow) * K + kk + scol,
                 &Bs[buf][srow][scol]);
  };

  stage(0, 0);
  const int nk = K >> 5;
  for (int ks = 0; ks < nk; ++ks) {
    const int buf = ks & 1;
    async_wait0();       // my async copies of `buf` (issued last iter) landed in LDS
    __syncthreads();     // everyone's landed
    if (ks + 1 < nk) stage(buf ^ 1, (ks + 1) << 5);

    // A fragment (16x32 bf16): lanes 0-15 hold K runs [0,8)+[16,24), lanes 16-31: [8,16)+[24,32)
    Frag a[2], b4[4];
#pragma unroll
    for (int ti = 0; ti < 2; ++ti) {
      const bf16* pa = &As[buf][wm + ti * 16 + lr][hi * 8];
      a[ti].q[0] = *(const uint4*)pa;
      a[ti].q[1] = *(const uint4*)(pa + 16);
    }
    // B fragment (32x16 bf16): lane n<16 = col n, K [0,8)+[16,24); lane n+16 = col n, K [8,16)+[24,32)
#pragma unroll
    for (int tj = 0; tj < 4; ++tj) {
      const bf16* pb = &Bs[buf][wn + tj * 16 + lr][hi * 8];
      b4[tj].q[0] = *(const uint4*)pb;
      b4[tj].q[1] = *(const uint4*)(pb + 16);
    }
#pragma unroll
    for (int ti = 0; ti < 2; ++ti)
#pragma unroll
      for (int tj = 0; tj < 4; ++tj)
        acc[ti][tj].v = __builtin_amdgcn_wmma_f32_16x16x32_bf16(
            false, a[ti].v, false, b4[tj].v, (short)0, acc[ti][tj].v, false, false);
  }

#pragma unroll
  for (int ti = 0; ti < 2; ++ti)
#pragma unroll
    for (int tj = 0; tj < 4; ++tj)
#pragma unroll
      for (int r = 0; r < 8; ++r) {
        const int gm = tileM + wm + ti * 16 + hi * 8 + r;
        const int gn = tileN + wn + tj * 16 + lr;
        const float val = acc[ti][tj].f[r] * scale;
        const size_t dix = doff + (size_t)gm * ldd + gn;
        if (dstB) dstB[dix] = (bf16)val;
        else      dstF[dix] = (resF ? resF[dix] : 0.f) + val;
      }
}

// ---------------------------------------------------------------------------
// 4) depthwise conv(3 along W) + bias + interleaved RoPE; pack [b,c,n,w,d] bf16
// ---------------------------------------------------------------------------
__global__ void qkpack_kernel(const float* __restrict__ pre,
                              const float* __restrict__ cw, const float* __restrict__ cb,
                              bf16* __restrict__ out) {
  int idx = blockIdx.x * blockDim.x + threadIdx.x;  // B*C*HEADS*(Dh/2)*Wn
  int w = idx % Wn; int t = idx / Wn;
  int i = t % (Dh / 2); t /= (Dh / 2);
  int n = t % HEADS; int bc = t / HEADS;
  if (bc >= Bn * Cc) return;
  int c = bc % Cc;
  const float* r0 = pre + ((size_t)bc * FEAT + n * Dh + 2 * i) * Wn;
  const float* r1 = r0 + Wn;
  float c0 = cw[c * 3 + 0], c1 = cw[c * 3 + 1], c2 = cw[c * 3 + 2], bb = cb[c];
  float l0 = (w > 0) ? r0[w - 1] : 0.f,  l1 = (w > 0) ? r1[w - 1] : 0.f;
  float p0 = (w < Wn - 1) ? r0[w + 1] : 0.f, p1 = (w < Wn - 1) ? r1[w + 1] : 0.f;
  float y0 = c0 * l0 + c1 * r0[w] + c2 * p0 + bb;
  float y1 = c0 * l1 + c1 * r1[w] + c2 * p1 + bb;
  float inv = __expf(-(float)(2 * i) * (9.210340372f / (float)Dh));  // 10000^(-2i/d)
  float ang = (float)w * inv;
  float cs = __cosf(ang), sn = __sinf(ang);
  size_t o = ((size_t)(bc * HEADS + n) * Wn + w) * Dh + 2 * i;
  out[o]     = (bf16)(y0 * cs - y1 * sn);
  out[o + 1] = (bf16)(y1 * cs + y0 * sn);
}

// 5) v: conv + bias only, layout preserved ([b,c,n,d,w] == [b,c,f,w]) -> bf16
__global__ void vpack_kernel(const float* __restrict__ pre,
                             const float* __restrict__ cw, const float* __restrict__ cb,
                             bf16* __restrict__ out) {
  int idx = blockIdx.x * blockDim.x + threadIdx.x;  // B*C*FEAT*Wn
  int w = idx % Wn; size_t row = (size_t)(idx / Wn);
  if (row >= (size_t)Bn * Cc * FEAT) return;
  int c = (int)((row / FEAT) % Cc);
  const float* r = pre + row * Wn;
  float y = cw[c * 3 + 0] * ((w > 0) ? r[w - 1] : 0.f)
          + cw[c * 3 + 1] * r[w]
          + cw[c * 3 + 2] * ((w < Wn - 1) ? r[w + 1] : 0.f) + cb[c];
  out[row * Wn + w] = (bf16)y;
}

// 6) softmax in place over rows of 512 bf16; one wave per row
__global__ void softmax_inplace_kernel(bf16* __restrict__ buf, int rows) {
  int lane = threadIdx.x & 31, wv = threadIdx.x >> 5;
  int row = blockIdx.x * 8 + wv;
  if (row >= rows) return;
  bf16* p = buf + (size_t)row * Wn + lane * 16;
  float v[16]; float mx = -3.0e38f;
#pragma unroll
  for (int i = 0; i < 16; ++i) { v[i] = (float)p[i]; mx = fmaxf(mx, v[i]); }
#pragma unroll
  for (int m = 16; m >= 1; m >>= 1) mx = fmaxf(mx, __shfl_xor(mx, m));
  float s = 0.f;
#pragma unroll
  for (int i = 0; i < 16; ++i) { v[i] = __expf(v[i] - mx); s += v[i]; }
#pragma unroll
  for (int m = 16; m >= 1; m >>= 1) s += __shfl_xor(s, m);
  float inv = 1.f / s;
#pragma unroll
  for (int i = 0; i < 16; ++i) p[i] = (bf16)(v[i] * inv);
}

// 7) 8x8 channel mix + relu^2; hid[b,c,hh,w] -> h2t[b,d,w,hh] (K-contig for lin2)
__global__ void chanmix_kernel(const bf16* __restrict__ hid,
                               const float* __restrict__ dwm,
                               bf16* __restrict__ out) {
  int idx = blockIdx.x * blockDim.x + threadIdx.x;  // Bn*EXPF*Wn
  int w = idx % Wn; int t = idx / Wn;
  int hh = t % EXPF; int b = t / EXPF;
  if (b >= Bn) return;
  float h[8];
#pragma unroll
  for (int c = 0; c < Cc; ++c)
    h[c] = (float)hid[(((size_t)b * Cc + c) * EXPF + hh) * Wn + w];
#pragma unroll
  for (int d = 0; d < Cc; ++d) {
    float s = 0.f;
#pragma unroll
    for (int c = 0; c < Cc; ++c) s += h[c] * dwm[d * Cc + c];
    float r = fmaxf(s, 0.f);
    out[(((size_t)b * Cc + d) * Wn + w) * (size_t)EXPF + hh] = (bf16)(r * r);
  }
}

// ---------------------------------------------------------------------------
extern "C" void kernel_launch(void* const* d_in, const int* in_sizes, int n_in,
                              void* d_out, int out_size, void* d_ws, size_t ws_size,
                              hipStream_t stream) {
  (void)in_sizes; (void)n_in; (void)out_size; (void)ws_size;
  const float* x       = (const float*)d_in[0];
  const float* embed_w = (const float*)d_in[1];
  const float* embed_b = (const float*)d_in[2];
  const float* norm1_w = (const float*)d_in[3];
  const float* norm1_b = (const float*)d_in[4];
  const float* q_pw = (const float*)d_in[5];
  const float* q_cw = (const float*)d_in[6];
  const float* q_cb = (const float*)d_in[7];
  const float* k_pw = (const float*)d_in[8];
  const float* k_cw = (const float*)d_in[9];
  const float* k_cb = (const float*)d_in[10];
  const float* v_pw = (const float*)d_in[11];
  const float* v_cw = (const float*)d_in[12];
  const float* v_cb = (const float*)d_in[13];
  const float* out_pw  = (const float*)d_in[14];
  const float* norm2_w = (const float*)d_in[15];
  const float* norm2_b = (const float*)d_in[16];
  const float* lin1_pw = (const float*)d_in[17];
  const float* lin1_dw = (const float*)d_in[18];
  const float* lin2_pw = (const float*)d_in[19];
  float* out = (float*)d_out;

  // workspace layout (256 MB, regions reused across phases)
  char* ws = (char*)d_ws;
  float* xbuf = (float*)(ws);                       // 33.5 MB f32, persistent (residual)
  bf16*  nt   = (bf16*)(ws + 33554432);             // 16.8 MB bf16, [b,c,w,f] (norm1 then norm2)
  char*  big  = ws + 50331648;                      // 100.7 MB shared region
  float* qpre = (float*)(big);                      //   q_pre f32
  float* kpre = (float*)(big + 33554432);           //   k_pre f32
  float* vpre = (float*)(big + 67108864);           //   v_pre f32
  bf16*  scor = (bf16*)(big);                       //   scores/P bf16 (after packs)
  bf16*  atp  = (bf16*)(big + 67108864);            //   attn out [b,c,w,f] bf16
  bf16*  hid  = (bf16*)(big);                       //   FFN hidden bf16 (after attn)
  char*  pk   = ws + 150994944;
  bf16*  qpk  = (bf16*)(pk);                        // 16.8 MB [b,c,n,w,d]
  bf16*  kpk  = (bf16*)(pk + 16777216);             // 16.8 MB
  bf16*  vt   = (bf16*)(pk + 33554432);             // 16.8 MB [b,c,n,d,w]
  bf16*  h2t  = (bf16*)(ws + 201326592);            // 67.1 MB [b,d,w,hh]

  auto gemm = [&](const void* A, int a_f32, const bf16* Bm,
                  float* dF, const float* rF, bf16* dB,
                  int M, int N, int K, long long aS, int aM, long long bS,
                  long long dO, int dDiv, long long dI, int ldd, float sc, int Z) {
    dim3 g(N / 128, M / 128, Z);
    gemm_nt_bf16_kernel<<<g, 256, 0, stream>>>(A, a_f32, Bm, dF, rF, dB,
        M, N, K, aS, aM, bS, dO, dDiv, dI, ldd, sc);
  };

  // 1) embed
  embed_kernel<<<(Bn * FEAT * Wn) / 256, 256, 0, stream>>>(x, embed_w, embed_b, xbuf);
  // 2) norm1 -> nt
  framenorm_t_kernel<<<(Bn * Cc * Wn) / 256, 256, 0, stream>>>(xbuf, norm1_w, norm1_b, nt);
  // 3) q/k/v per-channel projections (WMMA)
  gemm(q_pw, 1, nt, qpre, nullptr, nullptr, FEAT, Wn, FEAT,
       (long long)FEAT * FEAT, Cc, (long long)Wn * FEAT, (long long)FEAT * Wn, 1, 0, Wn, 1.f, Bn * Cc);
  gemm(k_pw, 1, nt, kpre, nullptr, nullptr, FEAT, Wn, FEAT,
       (long long)FEAT * FEAT, Cc, (long long)Wn * FEAT, (long long)FEAT * Wn, 1, 0, Wn, 1.f, Bn * Cc);
  gemm(v_pw, 1, nt, vpre, nullptr, nullptr, FEAT, Wn, FEAT,
       (long long)FEAT * FEAT, Cc, (long long)Wn * FEAT, (long long)FEAT * Wn, 1, 0, Wn, 1.f, Bn * Cc);
  // 4) dwconv + rope + pack
  int qkThreads = Bn * Cc * HEADS * (Dh / 2) * Wn;
  qkpack_kernel<<<qkThreads / 256, 256, 0, stream>>>(qpre, q_cw, q_cb, qpk);
  qkpack_kernel<<<qkThreads / 256, 256, 0, stream>>>(kpre, k_cw, k_cb, kpk);
  vpack_kernel<<<(Bn * Cc * FEAT * Wn) / 256, 256, 0, stream>>>(vpre, v_cw, v_cb, vt);
  // 5) scores = Q K^T / sqrt(FEAT)  (bf16 out), per head
  gemm(qpk, 0, kpk, nullptr, nullptr, scor, Wn, Wn, Dh,
       (long long)Wn * Dh, 0, (long long)Wn * Dh, (long long)Wn * Wn, 1, 0, Wn, 0.03125f, Bn * Cc * HEADS);
  // 6) softmax in place
  int rows = Bn * Cc * HEADS * Wn;
  softmax_inplace_kernel<<<rows / 8, 256, 0, stream>>>(scor, rows);
  // 7) A = P @ V -> atp[b,c,w, n*128+d]
  gemm(scor, 0, vt, nullptr, nullptr, atp, Wn, Dh, Wn,
       (long long)Wn * Wn, 0, (long long)Dh * Wn,
       (long long)Wn * FEAT, HEADS, (long long)Dh, FEAT, 1.f, Bn * Cc * HEADS);
  // 8) out projection + residual into xbuf (in place)
  gemm(out_pw, 1, atp, xbuf, xbuf, nullptr, FEAT, Wn, FEAT,
       (long long)FEAT * FEAT, Cc, (long long)Wn * FEAT, (long long)FEAT * Wn, 1, 0, Wn, 1.f, Bn * Cc);
  // 9) norm2 -> nt
  framenorm_t_kernel<<<(Bn * Cc * Wn) / 256, 256, 0, stream>>>(xbuf, norm2_w, norm2_b, nt);
  // 10) lin1 -> hid (bf16)
  gemm(lin1_pw, 1, nt, nullptr, nullptr, hid, EXPF, Wn, FEAT,
       (long long)EXPF * FEAT, Cc, (long long)Wn * FEAT, (long long)EXPF * Wn, 1, 0, Wn, 1.f, Bn * Cc);
  // 11) channel mix + relu^2 -> h2t
  chanmix_kernel<<<(Bn * EXPF * Wn) / 256, 256, 0, stream>>>(hid, lin1_dw, h2t);
  // 12) lin2 + residual -> d_out
  gemm(lin2_pw, 1, h2t, out, xbuf, nullptr, FEAT, Wn, EXPF,
       (long long)FEAT * EXPF, Cc, (long long)Wn * EXPF, (long long)FEAT * Wn, 1, 0, Wn, 1.f, Bn * Cc);
}
